// RelativeMultiHeadAttention_49417893708241
// MI455X (gfx1250) — matrix operand order repaired and checked by reading, not compiled, not run on hardware
//
#include <hip/hip_runtime.h>
#include <hip/hip_bf16.h>

// ---------------------------------------------------------------------------
// Relative Multi-Head Attention, CDNA5 (gfx1250), wave32 + WMMA f16->f32
// ---------------------------------------------------------------------------
typedef _Float16 v16h __attribute__((ext_vector_type(16)));
typedef _Float16 v8h  __attribute__((ext_vector_type(8)));
typedef float    v8f  __attribute__((ext_vector_type(8)));

#define D_MODEL 1024
#define NHEAD   16
#define DK      64
#define MAXREL  32
#define VOCAB   65             // 2*MAXREL+1
#define BATCH   4
#define TLEN    1024
#define MROWS   (BATCH*TLEN)   // 4096
#define QREL_S  80             // padded row stride of Qrel
#define RVK     96             // rel-value WMMA K (VOCAB padded to 3*32)
#define WB_S    104            // padded LDS stride for W rows (16B-aligned chunks)

static __device__ inline v8f vzero8() {
  v8f z = {0.f,0.f,0.f,0.f,0.f,0.f,0.f,0.f};
  return z;
}

// ---------------------------------------------------------------------------
// fp32 -> fp16 elementwise convert
// ---------------------------------------------------------------------------
__global__ void cvt_f32_f16(const float* __restrict__ src,
                            _Float16* __restrict__ dst, int n) {
  int i = blockIdx.x * blockDim.x + threadIdx.x;
  if (i < n) dst[i] = (_Float16)src[i];
}

// rel_value_emb [VOCAB, DK] fp32  ->  relvT [DK, RVK] f16 (zero padded)
__global__ void build_relvT(const float* __restrict__ relv,
                            _Float16* __restrict__ relvT) {
  const int d = blockIdx.x;       // 0..63
  const int r = threadIdx.x;      // 0..127
  if (r < RVK)
    relvT[(size_t)d * RVK + r] = (r < VOCAB) ? (_Float16)relv[r * DK + d]
                                             : (_Float16)0.f;
}

// ---------------------------------------------------------------------------
// GEMM: out = (A[M,K] @ W[N,K]^T + bias) * scale
// block = 256 thr (8 waves), tile 128x128, wave tile 32x64 (2x4 WMMA accums)
// ---------------------------------------------------------------------------
enum { STORE_HEAD = 0, STORE_VT = 1, STORE_F32 = 2 };

template <int MODE>
__global__ __launch_bounds__(256)
void gemm_xwT(const _Float16* __restrict__ A,     // [M,K] row-major
              const _Float16* __restrict__ W,     // [N,K] row-major
              const float*    __restrict__ bias,  // [N]
              void*           __restrict__ out,
              int M, int N, int K, float scale) {
  const int lane = threadIdx.x & 31;
  const int wid  = threadIdx.x >> 5;
  const int wm   = wid >> 1;
  const int wn   = wid & 1;
  const int nl   = lane & 15;
  const int hi   = lane >> 4;
  const int row0 = blockIdx.y * 128 + wm * 32;
  const int col0 = blockIdx.x * 128 + wn * 64;

  v8f acc[2][4];
#pragma unroll
  for (int mt = 0; mt < 2; ++mt)
#pragma unroll
    for (int nt = 0; nt < 4; ++nt) acc[mt][nt] = vzero8();

  for (int k0 = 0; k0 < K; k0 += 32) {
    v16h a[2], b[4];
#pragma unroll
    for (int mt = 0; mt < 2; ++mt) {
      const _Float16* ap = A + (size_t)(row0 + mt * 16 + nl) * K + k0;
      __builtin_prefetch((const void*)(ap + 64), 0, 0);   // two k-tiles ahead
      v8h lo = *(const v8h*)(ap + hi * 16);
      v8h hv = *(const v8h*)(ap + hi * 16 + 8);
#pragma unroll
      for (int e = 0; e < 8; ++e) { a[mt][e] = lo[e]; a[mt][8 + e] = hv[e]; }
    }
#pragma unroll
    for (int nt = 0; nt < 4; ++nt)
      b[nt] = *(const v16h*)(W + (size_t)(col0 + nt * 16 + nl) * K + k0 + hi * 16);
#pragma unroll
    for (int mt = 0; mt < 2; ++mt)
#pragma unroll
      for (int nt = 0; nt < 4; ++nt)
        acc[mt][nt] = __builtin_amdgcn_wmma_f32_16x16x32_f16(
            false, a[mt], false, b[nt], (short)0, acc[mt][nt], false, false);
  }

#pragma unroll
  for (int mt = 0; mt < 2; ++mt)
#pragma unroll
    for (int nt = 0; nt < 4; ++nt) {
      const int col = col0 + nt * 16 + nl;
      const float bv = bias[col];
#pragma unroll
      for (int j = 0; j < 8; ++j) {
        const int row = row0 + mt * 16 + hi * 8 + j;
        const float v = (acc[mt][nt][j] + bv) * scale;
        if (MODE == STORE_F32) {
          ((float*)out)[(size_t)row * N + col] = v;
        } else {
          const int b_ = row >> 10, t = row & 1023;   // row = b*T + t
          const int h  = col >> 6,  d = col & 63;     // col = h*DK + d
          if (MODE == STORE_HEAD)
            ((_Float16*)out)[(((size_t)(b_ * NHEAD + h)) * TLEN + t) * DK + d] = (_Float16)v;
          else  // STORE_VT: [BH, DK, T]
            ((_Float16*)out)[(((size_t)(b_ * NHEAD + h)) * DK + d) * TLEN + t] = (_Float16)v;
        }
      }
    }
}

// ---------------------------------------------------------------------------
// Qrel[row, r] = Q[row,:] . rel_key_emb[r,:]   row in [0, BH*T), r in [0,65)
// ---------------------------------------------------------------------------
__global__ __launch_bounds__(128)
void qrel_kernel(const _Float16* __restrict__ Qh,    // [BH*T, DK]
                 const float*    __restrict__ relk,  // [VOCAB, DK]
                 float*          __restrict__ qrel)  // [BH*T, QREL_S]
{
  __shared__ float qs[DK];
  const int row = blockIdx.x;
  const int tid = threadIdx.x;
  if (tid < DK) qs[tid] = (float)Qh[(size_t)row * DK + tid];
  __syncthreads();
  if (tid < VOCAB) {
    const float* e = relk + tid * DK;
    float s = 0.f;
#pragma unroll 8
    for (int d = 0; d < DK; ++d) s += qs[d] * e[d];
    qrel[(size_t)row * QREL_S + tid] = s;
  }
}

// ---------------------------------------------------------------------------
// Fused flash-style attention with relative position terms.
// grid = (T/64, B*H); block = 128 thr (4 waves); wave owns 16 q rows.
// K arrives pre-scaled by 1/sqrt(DK); rel term uses unscaled Q via Qrel.
//
// Relative terms per k-tile:
//  far tiles (|k-q|>=32 for all elems): rel addend is per-row constant
//    Qrel[q,0] / Qrel[q,64]; bucket sum == tile row-sum (already computed).
//  band tiles (<=3 per wave): gather Qrel interior, stash RAW scores in LDS
//    (no alpha rescale needed), clipped elems feed register tail partials.
// Epilogue: W[16 x 96] (tails + exp(band)) @ relvT via WMMA.
// ---------------------------------------------------------------------------
__global__ __launch_bounds__(128)
void attn_kernel(const _Float16* __restrict__ Qh,    // [BH, T, DK]
                 const _Float16* __restrict__ Kh,    // [BH, T, DK] (scaled)
                 const _Float16* __restrict__ Vt,    // [BH, DK, T]
                 const float*    __restrict__ qrel,  // [BH*T, QREL_S]
                 const _Float16* __restrict__ relvT, // [DK, RVK] f16
                 _Float16*       __restrict__ ctx)   // [B*T, D_MODEL]
{
  const int bh   = blockIdx.y;
  const int qt   = blockIdx.x;
  const int wid  = threadIdx.x >> 5;
  const int lane = threadIdx.x & 31;
  const int nl   = lane & 15;
  const int hi   = lane >> 4;
  const int q0   = qt * 64 + wid * 16;   // wave's first q (global t)

  __shared__ alignas(16) _Float16 Pld[4][16][72];    // P staging (C -> A layout)
  __shared__ alignas(16) _Float16 WB[4][16][WB_S];   // bucket weights (f16, A-side)
  __shared__ float Sband[4][16][66];                 // raw scores, diagonal band

  // init band buffer to -inf (per-wave private region; wave-internal in-order)
  {
    float* sb = &Sband[wid][0][0];
    for (int i = lane; i < 16 * 66; i += 32) sb[i] = -3.0e38f;
  }

  // Q operand (A matrix, 16x32 per chain), resident for whole k loop
  v16h aq[2];
#pragma unroll
  for (int c = 0; c < 2; ++c) {
    const _Float16* ap = Qh + ((size_t)bh * TLEN + q0 + nl) * DK + c * 32;
    v8h lo = *(const v8h*)(ap + hi * 16);
    v8h hv = *(const v8h*)(ap + hi * 16 + 8);
#pragma unroll
    for (int e = 0; e < 8; ++e) { aq[c][e] = lo[e]; aq[c][8 + e] = hv[e]; }
  }

  // per-row state (rows m = hi*8 + j live in this lane-half)
  const float* qrbase = qrel + ((size_t)bh * TLEN + q0 + hi * 8) * QREL_S;
  float rowmax[8], rowsum[8], tail0[8], tail1[8], qr0[8], qr64[8];
  v8f accv[4];
#pragma unroll
  for (int j = 0; j < 8; ++j) {
    rowmax[j] = -3.0e38f; rowsum[j] = 0.f; tail0[j] = 0.f; tail1[j] = 0.f;
    qr0[j]  = qrbase[(size_t)j * QREL_S];
    qr64[j] = qrbase[(size_t)j * QREL_S + 2 * MAXREL];
  }
#pragma unroll
  for (int nt = 0; nt < 4; ++nt) accv[nt] = vzero8();

  for (int kt = 0; kt < 16; ++kt) {
    const int k0 = kt * 64;

    // ---- S = Q . K^T  (c-outer: 4 independent accumulator chains) ----
    v8f s[4];
#pragma unroll
    for (int i = 0; i < 4; ++i) s[i] = vzero8();
#pragma unroll
    for (int c = 0; c < 2; ++c)
#pragma unroll
      for (int i = 0; i < 4; ++i) {
        v16h bk = *(const v16h*)(Kh + ((size_t)bh * TLEN + k0 + i * 16 + nl) * DK
                                 + c * 32 + hi * 16);
        s[i] = __builtin_amdgcn_wmma_f32_16x16x32_f16(
            false, aq[c], false, bk, (short)0, s[i], false, false);
      }

    // ---- relative-key addend + tile row max ----
    const bool farL = (k0 + 63 <= q0 - 32);     // all d = k-q <= -32
    const bool farR = (k0 >= q0 + 15 + 32);     // all d >= 32
    const int  dbase = (k0 + nl) - (q0 + hi * 8);
    float tmax[8];
#pragma unroll
    for (int j = 0; j < 8; ++j) tmax[j] = -3.0e38f;

    if (farL | farR) {
#pragma unroll
      for (int j = 0; j < 8; ++j) {
        const float qr = farL ? qr0[j] : qr64[j];
#pragma unroll
        for (int i = 0; i < 4; ++i) {
          const float v = s[i][j] + qr;
          s[i][j] = v;
          tmax[j] = fmaxf(tmax[j], v);
        }
      }
    } else {  // band tile: gather + stash raw scores
#pragma unroll
      for (int i = 0; i < 4; ++i)
#pragma unroll
        for (int j = 0; j < 8; ++j) {
          const int d = dbase + i * 16 - j;
          int rc = d < -MAXREL ? -MAXREL : (d > MAXREL ? MAXREL : d);
          const float v = s[i][j] + qrbase[(size_t)j * QREL_S + rc + MAXREL];
          s[i][j] = v;
          if (d > -MAXREL && d < MAXREL) Sband[wid][hi * 8 + j][d + MAXREL] = v;
          tmax[j] = fmaxf(tmax[j], v);
        }
    }
#pragma unroll
    for (int j = 0; j < 8; ++j)
#pragma unroll
      for (int m = 1; m < 16; m <<= 1)
        tmax[j] = fmaxf(tmax[j], __shfl_xor(tmax[j], m, 32));

    // ---- online softmax ----
    float alpha[8];
#pragma unroll
    for (int j = 0; j < 8; ++j) {
      const float nm = fmaxf(rowmax[j], tmax[j]);
      alpha[j] = __expf(rowmax[j] - nm);
      rowmax[j] = nm;
    }
    float tloc[8];
#pragma unroll
    for (int j = 0; j < 8; ++j) tloc[j] = 0.f;
#pragma unroll
    for (int i = 0; i < 4; ++i)
#pragma unroll
      for (int j = 0; j < 8; ++j) {
        const float p = __expf(s[i][j] - rowmax[j]);
        s[i][j] = p;
        tloc[j] += p;
      }
#pragma unroll
    for (int j = 0; j < 8; ++j) {
      float tr = tloc[j];
#pragma unroll
      for (int m = 1; m < 16; m <<= 1) tr += __shfl_xor(tr, m, 32);
      rowsum[j] = rowsum[j] * alpha[j] + tr;
      tail0[j] *= alpha[j];
      tail1[j] *= alpha[j];
    }

    // ---- clipped-bucket tail partials (registers, lane-local) ----
    if (farL) {
#pragma unroll
      for (int j = 0; j < 8; ++j) tail0[j] += tloc[j];
    } else if (farR) {
#pragma unroll
      for (int j = 0; j < 8; ++j) tail1[j] += tloc[j];
    } else {
#pragma unroll
      for (int i = 0; i < 4; ++i)
#pragma unroll
        for (int j = 0; j < 8; ++j) {
          const int d = dbase + i * 16 - j;
          if (d <= -MAXREL)     tail0[j] += s[i][j];
          else if (d >= MAXREL) tail1[j] += s[i][j];
        }
    }

#pragma unroll
    for (int nt = 0; nt < 4; ++nt)
#pragma unroll
      for (int j = 0; j < 8; ++j) accv[nt][j] *= alpha[j];

    // ---- P (C layout) -> LDS -> A layout ----
#pragma unroll
    for (int i = 0; i < 4; ++i)
#pragma unroll
      for (int j = 0; j < 8; ++j)
        Pld[wid][hi * 8 + j][i * 16 + nl] = (_Float16)s[i][j];

    v16h apm[2];
#pragma unroll
    for (int c = 0; c < 2; ++c) {
      const _Float16* pp = &Pld[wid][nl][c * 32];
      v8h lo = *(const v8h*)(pp + hi * 16);
      v8h hv = *(const v8h*)(pp + hi * 16 + 8);
#pragma unroll
      for (int e = 0; e < 8; ++e) { apm[c][e] = lo[e]; apm[c][8 + e] = hv[e]; }
    }

    // ---- accv += P . V  (c-outer: 4 independent chains) ----
#pragma unroll
    for (int c = 0; c < 2; ++c)
#pragma unroll
      for (int nt = 0; nt < 4; ++nt) {
        v16h bv = *(const v16h*)(Vt + ((size_t)bh * DK + nt * 16 + nl) * TLEN
                                 + k0 + c * 32 + hi * 16);
        accv[nt] = __builtin_amdgcn_wmma_f32_16x16x32_f16(
            false, apm[c], false, bv, (short)0, accv[nt], false, false);
      }
  }

  // ---- reduce tails across the 16 lanes of each half ----
#pragma unroll
  for (int j = 0; j < 8; ++j) {
#pragma unroll
    for (int m = 1; m < 16; m <<= 1) {
      tail0[j] += __shfl_xor(tail0[j], m, 32);
      tail1[j] += __shfl_xor(tail1[j], m, 32);
    }
  }

  // ---- build W rows (f16) : r=0 tail0, r=64 tail1, interior exp(band) ----
#pragma unroll
  for (int j = 0; j < 8; ++j) {
    const int row = hi * 8 + j;
    for (int e = nl; e < RVK; e += 16) {
      float wv;
      if (e == 0)        wv = tail0[j];
      else if (e == 2 * MAXREL) wv = tail1[j];
      else if (e < 2 * MAXREL)  wv = __expf(Sband[wid][row][e] - rowmax[j]);
      else               wv = 0.f;
      WB[wid][row][e] = (_Float16)wv;
    }
  }

  // ---- accv += W @ relvT  (3 K-chains x 4 d-tiles, WMMA) ----
#pragma unroll
  for (int c = 0; c < 3; ++c) {
    v16h aw;
    const _Float16* pp = &WB[wid][nl][c * 32];
    v8h lo = *(const v8h*)(pp + hi * 16);
    v8h hv = *(const v8h*)(pp + hi * 16 + 8);
#pragma unroll
    for (int e = 0; e < 8; ++e) { aw[e] = lo[e]; aw[8 + e] = hv[e]; }
#pragma unroll
    for (int nt = 0; nt < 4; ++nt) {
      v16h bv = *(const v16h*)(relvT + (size_t)(nt * 16 + nl) * RVK
                               + c * 32 + hi * 16);
      accv[nt] = __builtin_amdgcn_wmma_f32_16x16x32_f16(
          false, aw, false, bv, (short)0, accv[nt], false, false);
    }
  }

  // ---- normalize, store context [B*T, D_MODEL] ----
  const int b_ = bh >> 4, h = bh & 15;
#pragma unroll
  for (int j = 0; j < 8; ++j) {
    const int qg = q0 + hi * 8 + j;
    const float inv = 1.0f / rowsum[j];
#pragma unroll
    for (int nt = 0; nt < 4; ++nt) {
      const int dl = nt * 16 + nl;
      ctx[((size_t)(b_ * TLEN) + qg) * D_MODEL + h * DK + dl] =
          (_Float16)(accv[nt][j] * inv);
    }
  }
}

// ---------------------------------------------------------------------------
// host launch
// ---------------------------------------------------------------------------
extern "C" void kernel_launch(void* const* d_in, const int* in_sizes, int n_in,
                              void* d_out, int out_size, void* d_ws, size_t ws_size,
                              hipStream_t stream) {
  const float* query = (const float*)d_in[0];
  const float* key   = (const float*)d_in[1];
  const float* value = (const float*)d_in[2];
  const float* wq = (const float*)d_in[3];
  const float* bq = (const float*)d_in[4];
  const float* wk = (const float*)d_in[5];
  const float* bk = (const float*)d_in[6];
  const float* wv = (const float*)d_in[7];
  const float* bv = (const float*)d_in[8];
  const float* wo = (const float*)d_in[9];
  const float* bo = (const float*)d_in[10];
  const float* rel_key_emb   = (const float*)d_in[11];
  const float* rel_value_emb = (const float*)d_in[12];

  char* w = (char*)d_ws;
  auto alloc = [&](size_t bytes) -> void* {
    void* p = (void*)w;
    w += (bytes + 255) & ~(size_t)255;
    return p;
  };
  const size_t actB = (size_t)MROWS * D_MODEL * sizeof(_Float16);   // 8 MB
  const size_t wgtB = (size_t)D_MODEL * D_MODEL * sizeof(_Float16); // 2 MB
  _Float16* xq_h = (_Float16*)alloc(actB);
  _Float16* xk_h = (_Float16*)alloc(actB);
  _Float16* xv_h = (_Float16*)alloc(actB);
  _Float16* wq_h = (_Float16*)alloc(wgtB);
  _Float16* wk_h = (_Float16*)alloc(wgtB);
  _Float16* wv_h = (_Float16*)alloc(wgtB);
  _Float16* wo_h = (_Float16*)alloc(wgtB);
  _Float16* Qh   = (_Float16*)alloc(actB);   // [BH, T, DK]
  _Float16* Kh   = (_Float16*)alloc(actB);   // [BH, T, DK] scaled 1/8
  _Float16* Vt   = (_Float16*)alloc(actB);   // [BH, DK, T]
  float*    qrel = (float*)alloc((size_t)BATCH * NHEAD * TLEN * QREL_S * sizeof(float));
  _Float16* ctx  = (_Float16*)alloc(actB);   // [B*T, D_MODEL]
  _Float16* relvT = (_Float16*)alloc((size_t)DK * RVK * sizeof(_Float16));

  const int nAct = MROWS * D_MODEL, nWgt = D_MODEL * D_MODEL;
  cvt_f32_f16<<<(nAct + 255) / 256, 256, 0, stream>>>(query, xq_h, nAct);
  cvt_f32_f16<<<(nAct + 255) / 256, 256, 0, stream>>>(key,   xk_h, nAct);
  cvt_f32_f16<<<(nAct + 255) / 256, 256, 0, stream>>>(value, xv_h, nAct);
  cvt_f32_f16<<<(nWgt + 255) / 256, 256, 0, stream>>>(wq, wq_h, nWgt);
  cvt_f32_f16<<<(nWgt + 255) / 256, 256, 0, stream>>>(wk, wk_h, nWgt);
  cvt_f32_f16<<<(nWgt + 255) / 256, 256, 0, stream>>>(wv, wv_h, nWgt);
  cvt_f32_f16<<<(nWgt + 255) / 256, 256, 0, stream>>>(wo, wo_h, nWgt);
  build_relvT<<<DK, 128, 0, stream>>>(rel_value_emb, relvT);

  dim3 gGemm(D_MODEL / 128, MROWS / 128);  // (8, 32)
  const float kscale = 0.125f;             // 1/sqrt(DK)
  gemm_xwT<STORE_HEAD><<<gGemm, 256, 0, stream>>>(xq_h, wq_h, bq, Qh,
                                                  MROWS, D_MODEL, D_MODEL, 1.0f);
  gemm_xwT<STORE_HEAD><<<gGemm, 256, 0, stream>>>(xk_h, wk_h, bk, Kh,
                                                  MROWS, D_MODEL, D_MODEL, kscale);
  gemm_xwT<STORE_VT><<<gGemm, 256, 0, stream>>>(xv_h, wv_h, bv, Vt,
                                                MROWS, D_MODEL, D_MODEL, 1.0f);

  qrel_kernel<<<BATCH * NHEAD * TLEN, 128, 0, stream>>>(Qh, rel_key_emb, qrel);

  dim3 gAttn(TLEN / 64, BATCH * NHEAD);    // (16, 64)
  attn_kernel<<<gAttn, 128, 0, stream>>>(Qh, Kh, Vt, qrel, relvT, ctx);

  gemm_xwT<STORE_F32><<<gGemm, 256, 0, stream>>>(ctx, wo_h, bo, d_out,
                                                 MROWS, D_MODEL, D_MODEL, 1.0f);
}
